// FastRCNN_51531017617940
// MI455X (gfx1250) — matrix operand order repaired and checked by reading, not compiled, not run on hardware
//
#include <hip/hip_runtime.h>
#include <hip/hip_bf16.h>

typedef __attribute__((ext_vector_type(16))) __bf16 v16bf;
typedef __attribute__((ext_vector_type(8)))  float  v8f;

struct alignas(16) U16x8 { unsigned short u[8]; };

static __device__ __forceinline__ unsigned short f2bf(float f) {
  unsigned u = __builtin_bit_cast(unsigned, f);
  unsigned r = u + 0x7FFFu + ((u >> 16) & 1u);   // round-to-nearest-even
  return (unsigned short)(r >> 16);
}
static __device__ __forceinline__ float bf2f(unsigned short h) {
  unsigned u = ((unsigned)h) << 16;
  return __builtin_bit_cast(float, u);
}

// ---------------- generic f32 -> bf16 convert ----------------
__global__ void k_f32_to_bf16(const float* __restrict__ src,
                              unsigned short* __restrict__ dst, int n) {
  int i = blockIdx.x * blockDim.x + threadIdx.x;
  if (i < n) dst[i] = f2bf(src[i]);
}

// ---------------- BN fold: bias' = beta + (b - mean) * gamma * rsqrt(var+eps) ----------------
__global__ void k_fold_bias(const float* __restrict__ b, const float* __restrict__ g,
                            const float* __restrict__ be, const float* __restrict__ m,
                            const float* __restrict__ v, float* __restrict__ dst, int C) {
  int i = blockIdx.x * blockDim.x + threadIdx.x;
  if (i < C) {
    float s = g[i] * rsqrtf(v[i] + 1e-5f);
    dst[i] = be[i] + (b[i] - m[i]) * s;
  }
}

// ---------------- Pack conv weights (BN-scaled) into WMMA A-fragment order ----------------
// dst layout: [CoutTile][Kchunk][lane 0..31][half 0..15] of bf16
// A 16x32 bf16 layout (ISA 7.12.2): lane holds row M = lane&15,
//   half h holds K = (h&7) + 16*(h>>3) + 8*(lane>>4)
__global__ void k_pack_w(const float* __restrict__ W, const float* __restrict__ g,
                         const float* __restrict__ v, unsigned short* __restrict__ dst,
                         int Cin, int Cout) {
  int idx = blockIdx.x * blockDim.x + threadIdx.x;
  int K9 = Cin * 9;
  int Kc = (K9 + 31) >> 5;
  int total = (Cout >> 4) * Kc * 512;
  if (idx >= total) return;
  int h    = idx & 15;
  int lane = (idx >> 4) & 31;
  int kc   = (idx >> 9) % Kc;
  int ct   = idx / (Kc * 512);
  int mrow = ct * 16 + (lane & 15);
  int k    = kc * 32 + (h & 7) + 16 * (h >> 3) + 8 * (lane >> 4);
  float val = 0.f;
  if (k < K9) {
    float s = g[mrow] * rsqrtf(v[mrow] + 1e-5f);
    val = W[(size_t)mrow * K9 + k] * s;   // W is [Cout][Cin*9], k = cin*9 + tap
  }
  dst[idx] = f2bf(val);
}

// ---------------- 3x3 SAME conv as implicit GEMM, WMMA bf16 -> f32, fused bias+ReLU --------
// Workgroup (8 waves) computes a (16*MT Cout) x (16*PTW pixels) tile.
// K is processed in 128-wide blocks (4 WMMA chunks). For each block the im2col panel is
// gathered cooperatively into a double-buffered LDS panel laid out [n][k] with a 136-half
// row stride (272B = 68 dwords -> conflict-free ds_load_b128 fragment reads). Each thread's
// K column (ek = tid&127) is constant, so k/9 decomposition happens once per block, and the
// pixel coordinate advances by an add-and-wrap walk (no divisions in the fill loop).
__global__ __launch_bounds__(256) void k_conv_wmma(
    const unsigned short* __restrict__ in, unsigned short* __restrict__ out,
    const unsigned short* __restrict__ wpack, const float* __restrict__ bias,
    int Cin, int Cout, int H, int W) {
  __shared__ alignas(32) unsigned short ldsB[2][32 * 136];

  const int tid  = threadIdx.x;
  const int lane = tid & 31;
  const int wave = tid >> 5;

  const int MT  = (Cout >= 128) ? 8 : 4;   // Cout subtiles per workgroup
  const int PTW = 8 / MT;                  // pixel subtiles per workgroup
  const int PG  = 16 * PTW;                // pixels per workgroup
  const int HW  = H * W;
  const int PTg = HW / PG;                 // exact for all VGG layer sizes
  const int CTg = Cout / (16 * MT);

  const int bidx = blockIdx.x;
  const int pg = bidx % PTg;
  const int cg = (bidx / PTg) % CTg;
  const int b  = bidx / (PTg * CTg);

  const int mt = wave / PTW;
  const int pt = wave % PTW;

  const int K9  = Cin * 9;
  const int Kc  = (K9 + 31) >> 5;          // 32-wide WMMA chunks
  const int KcB = (Kc + 3) >> 2;           // 128-wide LDS panel blocks

  const unsigned short* __restrict__ inb = in + (size_t)b * Cin * HW;
  const int ctile = cg * MT + mt;
  const unsigned short* __restrict__ wp = wpack + (size_t)ctile * Kc * 512;

  // fixed per-thread fill coordinates: K column ek, pixel row walk start (x0,y0)
  const int ek  = tid & 127;
  const int en0 = tid >> 7;
  const int gn0 = pg * PG + en0;
  const int x0  = gn0 % W;
  const int y0  = gn0 / W;
  const int fillIters = PG >> 1;           // each iter advances pixel index by 2

  auto fill = [&](int kb, int bufI) {
    unsigned short* panel = &ldsB[bufI][0];
    int k   = kb * 128 + ek;
    int cin = k / 9;
    int tap = k - cin * 9;
    int dy  = tap / 3 - 1;
    int dx  = tap - (tap / 3) * 3 - 1;
    const bool kvalid = (k < K9);
    const unsigned short* __restrict__ src = inb + (size_t)cin * HW;
    int x = x0, y = y0;
    int off = en0 * 136 + ek;
    for (int i = 0; i < fillIters; ++i) {
      unsigned short val = 0;
      if (kvalid) {
        int yy = y + dy, xx = x + dx;
        if ((unsigned)yy < (unsigned)H && (unsigned)xx < (unsigned)W)
          val = src[yy * W + xx];
      }
      panel[off] = val;
      off += 2 * 136;
      x += 2;
      if (x >= W) { x -= W; y += 1; }      // add-and-wrap pixel walk
    }
  };

  const int nloc = pt * 16 + (lane & 15);
  const int n    = pg * PG + nloc;                     // this lane's output pixel
  const int ldsOffBase = nloc * 136 + 8 * (lane >> 4);

  v8f acc = {};
  fill(0, 0);
  for (int kb = 0; kb < KcB; ++kb) {
    __syncthreads();                                    // panel kb&1 ready / prev readers done
    if (kb + 1 < KcB) fill(kb + 1, (kb + 1) & 1);       // overlap next gather with WMMAs
    const unsigned short* panel = &ldsB[kb & 1][0];
    const int kcBase = kb * 4;
#pragma unroll
    for (int j = 0; j < 4; ++j) {
      int kc = kcBase + j;
      if (kc >= Kc) break;                              // wave-uniform
      v16bf a = *(const v16bf*)(wp + (size_t)kc * 512 + lane * 16);
      union { v16bf v; U16x8 q[2]; } bfrag;
      const unsigned short* p = panel + ldsOffBase + j * 32;
      bfrag.q[0] = *(const U16x8*)(p);
      bfrag.q[1] = *(const U16x8*)(p + 16);
      acc = __builtin_amdgcn_wmma_f32_16x16x32_bf16(false, a, false, bfrag.v,
                                                    (short)0, acc, false, false);
    }
  }
  // C/D layout: VGPR r -> row M = r + 8*(lane>>4), col N = lane&15
  if (n < HW) {
    unsigned short* __restrict__ outb = out + (size_t)b * Cout * HW;
    const int mbase = ctile * 16 + (lane >> 4) * 8;
#pragma unroll
    for (int r = 0; r < 8; ++r) {
      int m = mbase + r;
      float val = acc[r] + bias[m];
      val = val > 0.f ? val : 0.f;
      outb[(size_t)m * HW + n] = f2bf(val);
    }
  }
}

// ---------------- 2x2 stride-2 maxpool on bf16 ----------------
__global__ void k_maxpool2(const unsigned short* __restrict__ in,
                           unsigned short* __restrict__ out, int BC, int Ho, int Wo) {
  int idx = blockIdx.x * blockDim.x + threadIdx.x;
  int total = BC * Ho * Wo;
  if (idx >= total) return;
  int x = idx % Wo;
  int y = (idx / Wo) % Ho;
  int c = idx / (Wo * Ho);
  int Wi = Wo * 2;
  const unsigned short* p = in + ((size_t)c * Ho * 2 + y * 2) * Wi + x * 2;
  float a = bf2f(p[0]), bq = bf2f(p[1]), cc = bf2f(p[Wi]), d = bf2f(p[Wi + 1]);
  out[idx] = f2bf(fmaxf(fmaxf(a, bq), fmaxf(cc, d)));
}

// ---------------- adaptive 7x7 ROI max pool (feat 2x512x14x14) ----------------
__global__ void k_roi_pool(const unsigned short* __restrict__ feat,
                           const int* __restrict__ boxes,
                           unsigned short* __restrict__ roif) {
  int idx = blockIdx.x * blockDim.x + threadIdx.x;
  const int total = 128 * 512 * 49;
  if (idx >= total) return;
  int px = idx % 7;
  int py = (idx / 7) % 7;
  int c  = (idx / 49) % 512;
  int r  = idx / (49 * 512);
  int b  = r >> 6;
  const int* bx = boxes + r * 4;
  int fx = bx[0] / 32, fy = bx[1] / 32;
  int fw = bx[2] / 32; if (fw < 1) fw = 1;
  int fh = bx[3] / 32; if (fh < 1) fh = 1;
  int ys = fy + (py * fh) / 7;
  int ye = fy + ((py + 1) * fh + 6) / 7;
  int xs = fx + (px * fw) / 7;
  int xe = fx + ((px + 1) * fw + 6) / 7;
  ys = min(max(ys, 0), 13); ye = min(max(ye, ys + 1), 14);
  xs = min(max(xs, 0), 13); xe = min(max(xe, xs + 1), 14);
  const unsigned short* f = feat + ((size_t)b * 512 + c) * 196;
  float m = -INFINITY;
  for (int yy = ys; yy < ye; ++yy)
    for (int xx = xs; xx < xe; ++xx)
      m = fmaxf(m, bf2f(f[yy * 14 + xx]));
  roif[(size_t)r * 25088 + (size_t)c * 49 + py * 7 + px] = f2bf(m);
}

// ---------------- FC1: [128,25088] x [N,25088]^T via WMMA, fused bias+ReLU ----------------
__global__ __launch_bounds__(256) void k_fc1_wmma(
    const unsigned short* __restrict__ A,   // [128,25088] bf16 (roi feats)
    const unsigned short* __restrict__ Bw,  // [N,25088]  bf16 (fc weight)
    const float* __restrict__ bias, float* __restrict__ Hout, int N) {
  const int K = 25088;
  const int lane = threadIdx.x & 31;
  const int wid  = blockIdx.x * 8 + (threadIdx.x >> 5);
  const int NT = N >> 4;
  if (wid >= 8 * NT) return;
  const int nt = wid % NT;
  const int mt = wid / NT;
  const int rowA = mt * 16 + (lane & 15);
  const int colN = nt * 16 + (lane & 15);
  const unsigned short* __restrict__ arow = A  + (size_t)rowA * K + 8  * (lane >> 4);
  const unsigned short* __restrict__ brow = Bw + (size_t)colN * K + 8  * (lane >> 4);
  v8f acc = {};
  for (int k0 = 0; k0 < K; k0 += 32) {
    union { v16bf v; U16x8 q[2]; } af, bf;
    af.q[0] = *(const U16x8*)(arow + k0);        // K halves 0..7
    af.q[1] = *(const U16x8*)(arow + k0 + 16);   // K halves 16..23 (+8 hi-lane shift)
    bf.q[0] = *(const U16x8*)(brow + k0);
    bf.q[1] = *(const U16x8*)(brow + k0 + 16);
    acc = __builtin_amdgcn_wmma_f32_16x16x32_bf16(false, af.v, false, bf.v,
                                                  (short)0, acc, false, false);
  }
  const int nn = nt * 16 + (lane & 15);
  const float bv = bias[nn];
#pragma unroll
  for (int r = 0; r < 8; ++r) {
    int m = mt * 16 + r + 8 * (lane >> 4);
    float val = acc[r] + bv;
    Hout[(size_t)m * N + nn] = val > 0.f ? val : 0.f;
  }
}

// ---------------- FC2: tiny dense layer, plain f32 ----------------
__global__ void k_fc2(const float* __restrict__ Hin, const float* __restrict__ W2,
                      const float* __restrict__ b2, float* __restrict__ out,
                      int Nh, int No) {
  int idx = blockIdx.x * blockDim.x + threadIdx.x;
  if (idx >= 128 * No) return;
  int n = idx % No, r = idx / No;
  float s = b2[n];
  for (int k = 0; k < Nh; ++k) s += Hin[(size_t)r * Nh + k] * W2[(size_t)n * Nh + k];
  out[idx] = s;
}

extern "C" void kernel_launch(void* const* d_in, const int* in_sizes, int n_in,
                              void* d_out, int out_size, void* d_ws, size_t ws_size,
                              hipStream_t stream) {
  (void)in_sizes; (void)n_in; (void)out_size; (void)ws_size;
  char* ws = (char*)d_ws;
  const int cfgC[13] = {64, 64, 128, 128, 256, 256, 256, 512, 512, 512, 512, 512, 512};

  // ---- workspace layout (256B aligned) ----
  size_t cursor = 0;
  auto alloc = [&](size_t bytes) {
    cursor = (cursor + 255) & ~(size_t)255;
    size_t o = cursor; cursor += bytes; return o;
  };
  const size_t bufElems = (size_t)2 * 64 * 448 * 448;     // largest activation (bf16)
  size_t offBuf0 = alloc(bufElems * 2);
  size_t offBuf1 = alloc(bufElems * 2);
  size_t offWp[13]; int KcA[13];
  {
    int cin = 3;
    for (int i = 0; i < 13; ++i) {
      int Kc = (cin * 9 + 31) >> 5;
      KcA[i] = Kc;
      offWp[i] = alloc((size_t)(cfgC[i] >> 4) * Kc * 512 * 2);
      cin = cfgC[i];
    }
  }
  size_t offBias[13];
  for (int i = 0; i < 13; ++i) offBias[i] = alloc((size_t)cfgC[i] * 4);
  size_t offFcW1c = alloc((size_t)32 * 25088 * 2);
  size_t offFcW1r = alloc((size_t)64 * 25088 * 2);
  size_t offRoi   = alloc((size_t)128 * 25088 * 2);
  size_t offHc    = alloc((size_t)128 * 32 * 4);
  size_t offHr    = alloc((size_t)128 * 64 * 4);

  unsigned short* buf[2] = {(unsigned short*)(ws + offBuf0),
                            (unsigned short*)(ws + offBuf1)};

  // ---- 1) image f32 -> bf16 ----
  {
    int n = 2 * 3 * 448 * 448;
    k_f32_to_bf16<<<(n + 255) / 256, 256, 0, stream>>>((const float*)d_in[0], buf[0], n);
  }

  // ---- 2) VGG16-BN feature stack ----
  int curb = 0, H = 448, cin = 3;
  for (int i = 0; i < 13; ++i) {
    int cout = cfgC[i];
    const float* Wc  = (const float*)d_in[1 + i * 6 + 0];
    const float* bc  = (const float*)d_in[1 + i * 6 + 1];
    const float* gam = (const float*)d_in[1 + i * 6 + 2];
    const float* bet = (const float*)d_in[1 + i * 6 + 3];
    const float* mea = (const float*)d_in[1 + i * 6 + 4];
    const float* var = (const float*)d_in[1 + i * 6 + 5];
    unsigned short* wp = (unsigned short*)(ws + offWp[i]);
    float* bfold = (float*)(ws + offBias[i]);

    int totalPack = (cout >> 4) * KcA[i] * 512;
    k_pack_w<<<(totalPack + 255) / 256, 256, 0, stream>>>(Wc, gam, var, wp, cin, cout);
    k_fold_bias<<<(cout + 255) / 256, 256, 0, stream>>>(bc, gam, bet, mea, var, bfold, cout);

    int MT  = (cout >= 128) ? 8 : 4;
    int PTW = 8 / MT;
    int CTg = cout / (16 * MT);
    int PTg = (H * H) / (16 * PTW);         // exact for all layer sizes
    int blocks = CTg * PTg * 2;
    k_conv_wmma<<<blocks, 256, 0, stream>>>(buf[curb], buf[1 - curb], wp, bfold,
                                            cin, cout, H, H);
    curb = 1 - curb;

    if (i == 1 || i == 3 || i == 6 || i == 9 || i == 12) {
      int Ho = H >> 1;
      int totalp = 2 * cout * Ho * Ho;
      k_maxpool2<<<(totalp + 255) / 256, 256, 0, stream>>>(buf[curb], buf[1 - curb],
                                                           2 * cout, Ho, Ho);
      curb = 1 - curb;
      H = Ho;
    }
    cin = cout;
  }
  // feature map now in buf[curb]: [2,512,14,14] bf16

  // ---- 3) ROI pool ----
  unsigned short* roif = (unsigned short*)(ws + offRoi);
  {
    int total = 128 * 512 * 49;
    k_roi_pool<<<(total + 255) / 256, 256, 0, stream>>>(buf[curb], (const int*)d_in[87], roif);
  }

  // ---- 4) FC weights -> bf16 ----
  { int n = 32 * 25088;
    k_f32_to_bf16<<<(n + 255) / 256, 256, 0, stream>>>((const float*)d_in[79],
        (unsigned short*)(ws + offFcW1c), n); }
  { int n = 64 * 25088;
    k_f32_to_bf16<<<(n + 255) / 256, 256, 0, stream>>>((const float*)d_in[83],
        (unsigned short*)(ws + offFcW1r), n); }

  // ---- 5) FC1 via WMMA (fused bias+ReLU) ----
  k_fc1_wmma<<<(8 * 2 + 7) / 8, 256, 0, stream>>>(roif, (const unsigned short*)(ws + offFcW1c),
      (const float*)d_in[80], (float*)(ws + offHc), 32);
  k_fc1_wmma<<<(8 * 4 + 7) / 8, 256, 0, stream>>>(roif, (const unsigned short*)(ws + offFcW1r),
      (const float*)d_in[84], (float*)(ws + offHr), 64);

  // ---- 6) FC2 -> d_out (cls [128,21] then reg [128,4]) ----
  float* outp = (float*)d_out;
  { int total = 128 * 21;
    k_fc2<<<(total + 255) / 256, 256, 0, stream>>>((const float*)(ws + offHc),
        (const float*)d_in[81], (const float*)d_in[82], outp, 32, 21); }
  { int total = 128 * 4;
    k_fc2<<<(total + 255) / 256, 256, 0, stream>>>((const float*)(ws + offHr),
        (const float*)d_in[85], (const float*)d_in[86], outp + 128 * 21, 64, 4); }
}
